// NonLocalBlock_64879775973510
// MI455X (gfx1250) — compile-verified, matrix operand order checked by reading, not run on hardware
//
#include <hip/hip_runtime.h>

// ---------------------------------------------------------------------------
// Non-local block (attention) for MI455X / gfx1250, wave32 + WMMA bf16.
//   B=4, C=256, Ci=32, N=H*W=4096.
// Kernel 1: fused QKV 1x1-conv projection -> bf16 Q[n][i], K[n][i], Vt[o][n]
// Kernel 2: flash-attention (online softmax) computing S^T = K*Q^T so that
//           P^T (D-layout) is directly a legal WMMA B operand, and the
//           V^T * P^T output lands in [C][N] order => coalesced fused
//           BatchNorm + ReLU + residual epilogue. distance recomputed in VALU.
//           K/V tiles double-buffered in LDS via GLOBAL_LOAD_ASYNC_TO_LDS_B128
//           (ASYNCcnt) when the toolchain exposes the builtin.
// ---------------------------------------------------------------------------

typedef __attribute__((ext_vector_type(16))) __bf16 v16bf;
typedef __attribute__((ext_vector_type(2)))  __bf16 bf16x2;
typedef __attribute__((ext_vector_type(8)))  float  v8f;
typedef int v4i __attribute__((vector_size(16)));   // matches builtin param

#define B_   4
#define C_   256
#define CI_  32
#define N_   4096

#define LDS_AS __attribute__((address_space(3)))
#define GLB_AS __attribute__((address_space(1)))

#if __has_builtin(__builtin_amdgcn_global_load_async_to_lds_b128)
#define HAS_ASYNC_LDS 1
#else
#define HAS_ASYNC_LDS 0
#endif

// 16-bit WMMA A/B operand K-slot mapping (CDNA5 ISA 7.12.2):
// VGPR v (packed pair), lane-half h: pair base K = (v>=4?16:0) + 8h + 2*(v&3)
__device__ __forceinline__ int kpair_base(int v, int h) {
  return ((v & 4) << 2) + (h << 3) + ((v & 3) << 1);
}

// ============================ Kernel 1: QKV =================================
// Grid: B*(N/16) blocks of 128 threads (4 waves). Block computes a 16-column
// (n) slab of P = [wq;wk;wv] @ X  (320 x 16). Wave w owns 5 of the 20 j-tiles.
__global__ __launch_bounds__(128, 1)
void qkv_proj_kernel(const float* __restrict__ x,
                     const float* __restrict__ wv,
                     const float* __restrict__ wq,
                     const float* __restrict__ wk,
                     __bf16* __restrict__ Qb,   // [B][N][Ci]
                     __bf16* __restrict__ Kb,   // [B][N][Ci]
                     __bf16* __restrict__ Vb) { // [B][C][N]  (V transposed)
  __shared__ __bf16 XT[16 * 32];                // [n][c] bf16 tile

  const int tid  = threadIdx.x;
  const int lane = tid & 31;
  const int wave = tid >> 5;
  const int col  = lane & 15;   // n (B-operand) / j (A-operand row) in tile
  const int h    = lane >> 4;
  const int b    = blockIdx.x >> 8;
  const int n0   = (blockIdx.x & 255) << 4;

  const v8f zero8 = {0.f, 0.f, 0.f, 0.f, 0.f, 0.f, 0.f, 0.f};
  v8f acc[5];
#pragma unroll
  for (int t = 0; t < 5; ++t) acc[t] = zero8;

  for (int ks = 0; ks < 8; ++ks) {            // K loop over C=256, step 32
    const int c0 = ks * 32;
    __syncthreads();
    { // stage X^T tile: coalesced 16B global loads, bf16 convert into LDS
      const int c  = tid >> 2;
      const int n4 = (tid & 3) << 2;
      const float4 xv =
          *(const float4*)(x + (size_t)(b * C_ + c0 + c) * N_ + n0 + n4);
      XT[(n4 + 0) * 32 + c] = (__bf16)xv.x;
      XT[(n4 + 1) * 32 + c] = (__bf16)xv.y;
      XT[(n4 + 2) * 32 + c] = (__bf16)xv.z;
      XT[(n4 + 3) * 32 + c] = (__bf16)xv.w;
    }
    __syncthreads();

    v16bf bX;                                  // B operand: X[c][n]
#pragma unroll
    for (int v = 0; v < 8; ++v) {
      const int kb = kpair_base(v, h);
      const bf16x2 p = *(const bf16x2*)&XT[col * 32 + kb];
      bX[2 * v] = p[0];
      bX[2 * v + 1] = p[1];
    }

#pragma unroll
    for (int t = 0; t < 5; ++t) {
      const int jt = wave * 5 + t;
      const int j  = jt * 16 + col;            // A-operand row (output channel)
      const float* wrow = (jt < 2) ? (wq + (size_t)j * C_)
                        : (jt < 4) ? (wk + (size_t)(j - CI_) * C_)
                                   : (wv + (size_t)(j - 2 * CI_) * C_);
      v16bf aW;                                // A operand: W[j][c]
#pragma unroll
      for (int v = 0; v < 8; ++v) {
        const int kb = kpair_base(v, h);
        const float2 w2 = *(const float2*)(wrow + c0 + kb);
        aW[2 * v] = (__bf16)w2.x;
        aW[2 * v + 1] = (__bf16)w2.y;
      }
      acc[t] = __builtin_amdgcn_wmma_f32_16x16x32_bf16(
          false, aW, false, bX, (short)0, acc[t], false, false);
    }
  }

  // ---- store: Q/K as [n][i] rows (one 16B store/lane), V as [o][n] ----
#pragma unroll
  for (int t = 0; t < 5; ++t) {
    const int jt = wave * 5 + t;
    if (jt < 4) {
      alignas(16) __bf16 pk[8];
#pragma unroll
      for (int v = 0; v < 8; ++v) pk[v] = (__bf16)acc[t][v];
      __bf16* dst =
          (jt < 2)
              ? (Qb + (size_t)(b * N_ + n0 + col) * CI_ + (jt * 16 + h * 8))
              : (Kb + (size_t)(b * N_ + n0 + col) * CI_ +
                 ((jt - 2) * 16 + h * 8));
      *(int4*)dst = *(const int4*)pk;
    } else {
#pragma unroll
      for (int v = 0; v < 8; ++v) {
        const int o = (jt - 4) * 16 + v + 8 * h;
        Vb[(size_t)(b * C_ + o) * N_ + n0 + col] = (__bf16)acc[t][v];
      }
    }
  }
}

// ====================== Kernel 2: flash attention ===========================
// Per-32-key-step staging of K (32x32, 2 KB) and V^T (256x32, 16 KB) tiles,
// double-buffered; async global->LDS DMA when available (9 b128 per thread).
__device__ __forceinline__ void stage_kv(const __bf16* __restrict__ Kbase,
                                         const __bf16* __restrict__ Vbase,
                                         __bf16* kt, __bf16* vt, int tid) {
  const int m = tid >> 2;
  const int part = (tid & 3) << 3;
#if HAS_ASYNC_LDS
  __builtin_amdgcn_global_load_async_to_lds_b128(
      (GLB_AS v4i*)(Kbase + m * CI_ + part),
      (LDS_AS v4i*)(kt + m * CI_ + part), 0, 0);
#pragma unroll
  for (int r = 0; r < 2; ++r) {
    const int o = tid * 2 + r;
#pragma unroll
    for (int j = 0; j < 4; ++j)
      __builtin_amdgcn_global_load_async_to_lds_b128(
          (GLB_AS v4i*)(Vbase + (size_t)o * N_ + j * 8),
          (LDS_AS v4i*)(vt + o * CI_ + j * 8), 0, 0);
  }
#else
  *(int4*)(kt + m * CI_ + part) = *(const int4*)(Kbase + m * CI_ + part);
#pragma unroll
  for (int r = 0; r < 2; ++r) {
    const int o = tid * 2 + r;
    const int4* s4 = (const int4*)(Vbase + (size_t)o * N_);
    int4* dst = (int4*)(vt + o * CI_);
    dst[0] = s4[0]; dst[1] = s4[1]; dst[2] = s4[2]; dst[3] = s4[3];
  }
#endif
}

__device__ __forceinline__ void wait_async_lds() {
#if HAS_ASYNC_LDS
#if __has_builtin(__builtin_amdgcn_s_wait_asynccnt)
  __builtin_amdgcn_s_wait_asynccnt(0);
#else
  asm volatile("s_wait_asynccnt 0" ::: "memory");
#endif
#endif
}

// Grid: B*(N/64) blocks of 128 threads. Each wave owns 16 query rows; the 4
// waves share the LDS-staged K / V^T tiles.
__global__ __launch_bounds__(128, 1)
void attn_bn_kernel(const __bf16* __restrict__ Qb,
                    const __bf16* __restrict__ Kb,
                    const __bf16* __restrict__ Vb,
                    const float* __restrict__ x,
                    const float* __restrict__ gamma,
                    const float* __restrict__ beta,
                    const float* __restrict__ mean,
                    const float* __restrict__ var,
                    float* __restrict__ out) {
  __shared__ __bf16 Kt[2][32 * 32];    // [buf][m_local][i]
  __shared__ __bf16 Vt[2][256 * 32];   // [buf][o][m_local]

  const int tid  = threadIdx.x;
  const int lane = tid & 31;
  const int wave = tid >> 5;
  const int col  = lane & 15;
  const int h    = lane >> 4;
  const int b    = blockIdx.x >> 6;
  const int n0   = (blockIdx.x & 63) << 6;
  const int nq   = n0 + wave * 16 + col;       // this lane's query row
  const float qy = (float)(nq >> 6);
  const float qx = (float)(nq & 63);

  const __bf16* Kbatch = Kb + (size_t)b * N_ * CI_;
  const __bf16* Vbatch = Vb + (size_t)b * C_ * N_;

  // Q^T B-operand (persistent): B[i][q] = Qb[nq][i]
  v16bf bQ;
  {
    const __bf16* qrow = Qb + (size_t)(b * N_ + nq) * CI_;
#pragma unroll
    for (int v = 0; v < 8; ++v) {
      const int kb = kpair_base(v, h);
      const bf16x2 p = *(const bf16x2*)(qrow + kb);
      bQ[2 * v] = p[0];
      bQ[2 * v + 1] = p[1];
    }
  }

  const v8f zero8 = {0.f, 0.f, 0.f, 0.f, 0.f, 0.f, 0.f, 0.f};
  v8f acc[16];                                  // y^T[o][q], 128 VGPRs
#pragma unroll
  for (int t = 0; t < 16; ++t) acc[t] = zero8;

  float rm = -3.0e38f;                          // running row max (row == nq)
  float rl = 0.0f;                              // running row sum

  // Prologue: stage tile 0 into buffer 0.
  stage_kv(Kbatch, Vbatch, Kt[0], Vt[0], tid);

  for (int m0 = 0; m0 < N_; m0 += 32) {
    const int buf = (m0 >> 5) & 1;
    wait_async_lds();          // our async copies for `buf` are done
    __syncthreads();           // everyone's copies are done & visible
    if (m0 + 32 < N_)          // overlap next tile DMA with this tile's WMMAs
      stage_kv(Kbatch + (size_t)(m0 + 32) * CI_, Vbatch + (m0 + 32),
               Kt[buf ^ 1], Vt[buf ^ 1], tid);

    const __bf16* kt = Kt[buf];
    const __bf16* vt = Vt[buf];

    // ---- S^T = K * Q^T : two 16x16 tiles over 32 keys (K-dim = Ci = 32) ----
    v16bf aK0, aK1;
#pragma unroll
    for (int v = 0; v < 8; ++v) {
      const int kb = kpair_base(v, h);
      const bf16x2 p0 = *(const bf16x2*)&kt[col * 32 + kb];
      const bf16x2 p1 = *(const bf16x2*)&kt[(col + 16) * 32 + kb];
      aK0[2 * v] = p0[0]; aK0[2 * v + 1] = p0[1];
      aK1[2 * v] = p1[0]; aK1[2 * v + 1] = p1[1];
    }
    v8f s0 = __builtin_amdgcn_wmma_f32_16x16x32_bf16(
        false, aK0, false, bQ, (short)0, zero8, false, false);
    v8f s1 = __builtin_amdgcn_wmma_f32_16x16x32_bf16(
        false, aK1, false, bQ, (short)0, zero8, false, false);

    // ---- 1/distance scaling (recomputed from grid coords) + online softmax
    float p0a[8], p1a[8];
    float tmax = -3.0e38f;
#pragma unroll
    for (int v = 0; v < 8; ++v) {
      int mg = m0 + v + 8 * h;
      float dy = (float)(mg >> 6) - qy;
      float dx = (float)(mg & 63) - qx;
      float s = s0[v] / (sqrtf(dx * dx + dy * dy) + 1.0f);
      p0a[v] = s;
      tmax = fmaxf(tmax, s);
      mg += 16;
      dy = (float)(mg >> 6) - qy;
      dx = (float)(mg & 63) - qx;
      s = s1[v] / (sqrtf(dx * dx + dy * dy) + 1.0f);
      p1a[v] = s;
      tmax = fmaxf(tmax, s);
    }
    tmax = fmaxf(tmax, __shfl_xor(tmax, 16, 32)); // row lives on lanes j,j+16
    const float nm = fmaxf(rm, tmax);
    const float corr = __expf(rm - nm);
    float lsum = 0.0f;
#pragma unroll
    for (int v = 0; v < 8; ++v) {
      p0a[v] = __expf(p0a[v] - nm);
      p1a[v] = __expf(p1a[v] - nm);
      lsum += p0a[v] + p1a[v];
    }
    lsum += __shfl_xor(lsum, 16, 32);
    rl = rl * corr + lsum;
    rm = nm;

    // P^T in D-layout is directly a legal 32x16 bf16 B operand:
    // slots 0..7 <- tile0 vgprs 0..7, slots 8..15 <- tile1 vgprs 0..7.
    v16bf bP;
#pragma unroll
    for (int v = 0; v < 8; ++v) {
      bP[v] = (__bf16)p0a[v];
      bP[v + 8] = (__bf16)p1a[v];
    }

    // ---- acc^T += V^T * P^T over all 256 output channels ----
#pragma unroll
    for (int t = 0; t < 16; ++t) {
#pragma unroll
      for (int v = 0; v < 8; ++v) acc[t][v] *= corr; // flash rescale
      v16bf aV;                                       // A: V^T[o][m]
      const __bf16* vr = &vt[(t * 16 + col) * 32];
#pragma unroll
      for (int v = 0; v < 8; ++v) {
        const int kb = kpair_base(v, h);
        const bf16x2 p = *(const bf16x2*)(vr + kb);
        aV[2 * v] = p[0];
        aV[2 * v + 1] = p[1];
      }
      acc[t] = __builtin_amdgcn_wmma_f32_16x16x32_bf16(
          false, aV, false, bP, (short)0, acc[t], false, false);
    }
  }

  // ---- epilogue: softmax normalize + BN + ReLU + residual, coalesced ----
  const float inv_l = 1.0f / rl;
#pragma unroll
  for (int t = 0; t < 16; ++t) {
#pragma unroll
    for (int v = 0; v < 8; ++v) {
      const int o = t * 16 + v + 8 * h;
      const float sc = gamma[o] * rsqrtf(var[o] + 1e-5f);
      const float sh = beta[o] - mean[o] * sc;
      const size_t idx = (size_t)(b * C_ + o) * N_ + nq;
      const float y = acc[t][v] * inv_l * sc + sh;
      out[idx] = fmaxf(y, 0.0f) + x[idx];
    }
  }
}

// =============================== launch =====================================
extern "C" void kernel_launch(void* const* d_in, const int* in_sizes, int n_in,
                              void* d_out, int out_size, void* d_ws,
                              size_t ws_size, hipStream_t stream) {
  const float* x     = (const float*)d_in[0];
  const float* wv    = (const float*)d_in[1];
  const float* wq    = (const float*)d_in[2];
  const float* wk    = (const float*)d_in[3];
  const float* gamma = (const float*)d_in[4];
  const float* beta  = (const float*)d_in[5];
  const float* mean  = (const float*)d_in[6];
  const float* var   = (const float*)d_in[7];
  // d_in[8] (distance) is recomputed on the fly in-kernel.
  float* out = (float*)d_out;

  __bf16* Qb = (__bf16*)d_ws;                    // [B][N][Ci] bf16
  __bf16* Kb = Qb + (size_t)B_ * N_ * CI_;       // [B][N][Ci] bf16
  __bf16* Vb = Kb + (size_t)B_ * N_ * CI_;       // [B][C][N]  bf16

  qkv_proj_kernel<<<B_ * (N_ / 16), 128, 0, stream>>>(x, wv, wq, wk, Qb, Kb,
                                                      Vb);
  attn_bn_kernel<<<B_ * (N_ / 64), 128, 0, stream>>>(Qb, Kb, Vb, x, gamma,
                                                     beta, mean, var, out);
}
